// LSTM_3977139716545
// MI455X (gfx1250) — compile-verified, matrix operand order checked by reading, not run on hardware
//
#include <hip/hip_runtime.h>
#include <hip/hip_bf16.h>
#include <math.h>

// ---------------------------------------------------------------------------
// 2-layer LSTM (B=256, T=2048, H=128, Cin=12) + dense sigmoid head, MI455X.
// 16 WGs x 256 threads; each WG owns a 16-row batch tile and scans all T steps
// with h/c resident on-chip. GEMMs via v_wmma_f32_16x16x32_bf16 (B-fragments
// in VGPRs, matrix_a_reuse hints), single barrier/step via ping-pong LDS,
// async global<->LDS staging (ASYNCcnt) for activation tiles, and native
// v_tanh_f32 / v_exp_f32-based gate nonlinearities.
// ---------------------------------------------------------------------------

typedef __attribute__((ext_vector_type(16))) __bf16 v16bf;
typedef __attribute__((ext_vector_type(8)))  float  v8f;

#define Bsz   256
#define Tlen  2048
#define Hdim  128
#define Cin   12
#define G4H   512
#define BT    16     // batch tile (WMMA M)
#define HSTR  136    // LDS row stride (halves) for h/y tiles: 272B, 16B aligned
#define XSTR  48     // LDS row stride (halves) for x tile (K padded 12->32)

#define LOG2E 1.44269504088896340736f

__device__ __forceinline__ float fast_tanh(float x) {
#if __has_builtin(__builtin_amdgcn_tanhf)
    return __builtin_amdgcn_tanhf(x);          // v_tanh_f32
#else
    float r;
    asm("v_tanh_f32 %0, %1" : "=v"(r) : "v"(x));
    return r;
#endif
}
__device__ __forceinline__ float fast_sigmoid(float x) {
    // 1 / (1 + 2^(-x*log2e)) : v_mul + v_exp_f32 + v_add + v_rcp_f32, branch-free
    return __builtin_amdgcn_rcpf(1.0f + __builtin_amdgcn_exp2f(-x * LOG2E));
}

__device__ __forceinline__ v8f wmma_bf16(v16bf a, v16bf b, v8f c) {
    return __builtin_amdgcn_wmma_f32_16x16x32_bf16(
        false, a, false, b, (short)0, c, /*reuse_a=*/false, /*reuse_b=*/false);
}
// Same-A follower: previous VALU-matrix op is identical and shares A -> RA hint.
__device__ __forceinline__ v8f wmma_bf16_ra(v16bf a, v16bf b, v8f c) {
    return __builtin_amdgcn_wmma_f32_16x16x32_bf16(
        false, a, false, b, (short)0, c, /*reuse_a=*/true, /*reuse_b=*/false);
}

// A-fragment (16M x 32K bf16) from a row-major LDS tile [16][stride_h].
// Lanes 0-15: M=lane, K=0..7 (e0..7), K=16..23 (e8..15); lanes 16-31: +8.
__device__ __forceinline__ v16bf load_a_frag(const __bf16* buf, int stride_h,
                                             int lane, int kbase) {
    int m    = lane & 15;
    int koff = kbase + ((lane >> 4) << 3);
    union { uint4 u[2]; v16bf v; } r;
    const __bf16* p = buf + m * stride_h + koff;
    r.u[0] = *(const uint4*)(p);
    r.u[1] = *(const uint4*)(p + 16);
    return r.v;
}

// B-fragment: packed contiguously per (frag_idx, lane): 16 halves = 32B.
__device__ __forceinline__ v16bf load_b_frag(const __bf16* base, int frag_idx, int lane) {
    return *(const v16bf*)(base + ((size_t)frag_idx * 32 + lane) * 16);
}

// Pack a [512 x 128] f32 weight matrix into bf16 WMMA B-fragments in LDS.
__device__ __forceinline__ void pack_weight_128(const float* __restrict__ W,
                                                __bf16* __restrict__ frag, int tid) {
    for (int idx = tid; idx < 32 * 4 * 32 * 16; idx += 256) {
        int e  = idx & 15;
        int ln = (idx >> 4) & 31;
        int kt = (idx >> 9) & 3;
        int nt = idx >> 11;
        int kk = (e & 7) | ((e & 8) << 1) | ((ln >> 4) << 3);
        int n  = (nt << 4) | (ln & 15);
        int k  = (kt << 5) | kk;
        frag[idx] = (__bf16)W[n * Hdim + k];
    }
}

// ------------------------------ layer 0 ------------------------------------
// x: [B, Cin, T] f32.  y out: [T, B, H] bf16 (workspace).
__global__ void __launch_bounds__(256, 1)
lstm_layer0(const float* __restrict__ x,
            const float* __restrict__ Wih, const float* __restrict__ Whh,
            const float* __restrict__ bih, const float* __restrict__ bhh,
            __bf16* __restrict__ yout) {
    extern __shared__ __align__(32) char smem[];
    __bf16* wih_f = (__bf16*)(smem);                                // 32768 B
    __bf16* whh_f = (__bf16*)(smem + 32768);                        // 131072 B
    float*  bias  = (float*)(smem + 32768 + 131072);                // 2048 B
    __bf16* hbuf  = (__bf16*)(smem + 32768 + 131072 + 2048);        // 2x 4352 B
    __bf16* xbuf  = (__bf16*)(smem + 32768 + 131072 + 2048 + 8704); // 2x 1536 B

    const int tid  = threadIdx.x;
    const int lane = tid & 31;
    const int w    = tid >> 5;          // wave id, owns hidden cols [16w,16w+16)
    const int nn   = lane & 15;
    const int hi   = lane >> 4;
    const int col  = (w << 4) + nn;
    const int b0   = blockIdx.x * BT;

    pack_weight_128(Whh, whh_f, tid);
    // W_ih0: K=12 padded to 32 (zeros), single K-tile per N-tile.
    for (int idx = tid; idx < 32 * 32 * 16; idx += 256) {
        int e  = idx & 15;
        int ln = (idx >> 4) & 31;
        int nt = idx >> 9;
        int kk = (e & 7) | ((e & 8) << 1) | ((ln >> 4) << 3);
        int n  = (nt << 4) | (ln & 15);
        wih_f[idx] = (kk < Cin) ? (__bf16)Wih[n * Cin + kk] : (__bf16)0.0f;
    }
    for (int i = tid; i < G4H; i += 256)            bias[i] = bih[i] + bhh[i];
    for (int i = tid; i < 2 * BT * HSTR; i += 256)  hbuf[i] = (__bf16)0.0f;
    for (int i = tid; i < 2 * BT * XSTR; i += 256)  xbuf[i] = (__bf16)0.0f;
    __syncthreads();

    // Hoist all loop-invariant B-fragments into VGPRs (20 x v16bf = 160 VGPRs).
    v16bf bI[4], bH[4][4];
#pragma unroll
    for (int g = 0; g < 4; ++g) {
        bI[g] = load_b_frag(wih_f, g * 8 + w, lane);
#pragma unroll
        for (int kt = 0; kt < 4; ++kt)
            bH[g][kt] = load_b_frag(whh_f, ((g * 8 + w) << 2) | kt, lane);
    }
    const float bi_i = bias[0 * Hdim + col];
    const float bi_f = bias[1 * Hdim + col];
    const float bi_g = bias[2 * Hdim + col];
    const float bi_o = bias[3 * Hdim + col];

    // staging assignments
    const int bl  = tid / Cin;          // x: thread -> (row, channel)
    const int cch = tid - bl * Cin;
    const int sr  = tid >> 4;           // y: thread -> (row, 8-half group)
    const int sc  = (tid & 15) << 3;

    // Prologue: stage x_0 into buffer 0.
    if (tid < BT * Cin)
        xbuf[bl * XSTR + cch] = (__bf16)x[((size_t)(b0 + bl) * Cin + cch) * Tlen];
    __syncthreads();

    v8f cst = {0.f, 0.f, 0.f, 0.f, 0.f, 0.f, 0.f, 0.f};

    for (int t = 0; t < Tlen; ++t) {
        const int cur = t & 1, nxt = cur ^ 1;
        const __bf16* xb_c = xbuf + cur * (BT * XSTR);
        __bf16*       xb_n = xbuf + nxt * (BT * XSTR);
        const __bf16* hb_c = hbuf + cur * (BT * HSTR);
        __bf16*       hb_n = hbuf + nxt * (BT * HSTR);

        // Async-store y[t-1] (= h_t, sitting in hb_c) to global, LDS->memory.
        // hb_c is only rewritten in step t+1, and we wait ASYNCcnt before then.
        if (t > 0) {
            uint32_t ldsa = (uint32_t)(uintptr_t)(hb_c + sr * HSTR + sc);
            const __bf16* gdst = yout + ((size_t)(t - 1) * Bsz + b0 + sr) * Hdim + sc;
            asm volatile("global_store_async_from_lds_b128 %0, %1, off"
                         :: "v"(gdst), "v"(ldsa) : "memory");
        }

        // Issue next-step x loads early (consumed after compute).
        float xv = 0.f;
        if (t + 1 < Tlen && tid < BT * Cin)
            xv = x[((size_t)(b0 + bl) * Cin + cch) * Tlen + (t + 1)];

        v8f z = {0.f, 0.f, 0.f, 0.f, 0.f, 0.f, 0.f, 0.f};
        v8f acc0 = z, acc1 = z, acc2 = z, acc3 = z;

        // input projection (one padded K-tile), B in registers
        {
            v16bf aX = load_a_frag(xb_c, XSTR, lane, 0);
            acc0 = wmma_bf16   (aX, bI[0], acc0);
            acc1 = wmma_bf16_ra(aX, bI[1], acc1);
            acc2 = wmma_bf16_ra(aX, bI[2], acc2);
            acc3 = wmma_bf16_ra(aX, bI[3], acc3);
        }
        // recurrent h @ Whh^T, B in registers
#pragma unroll
        for (int kt = 0; kt < 4; ++kt) {
            v16bf aH = load_a_frag(hb_c, HSTR, lane, kt << 5);
            acc0 = wmma_bf16   (aH, bH[0][kt], acc0);
            acc1 = wmma_bf16_ra(aH, bH[1][kt], acc1);
            acc2 = wmma_bf16_ra(aH, bH[2][kt], acc2);
            acc3 = wmma_bf16_ra(aH, bH[3][kt], acc3);
        }

        // gate nonlinearities + cell update; write h into the *other* buffer
#pragma unroll
        for (int r = 0; r < 8; ++r) {
            float gi = fast_sigmoid(acc0[r] + bi_i);
            float gf = fast_sigmoid(acc1[r] + bi_f);
            float gg = fast_tanh(acc2[r] + bi_g);
            float go = fast_sigmoid(acc3[r] + bi_o);
            float cn = gf * cst[r] + gi * gg;
            cst[r] = cn;
            float h = go * fast_tanh(cn);
            int M = r + (hi << 3);
            hb_n[M * HSTR + col] = (__bf16)h;
        }
        if (t + 1 < Tlen && tid < BT * Cin)
            xb_n[bl * XSTR + cch] = (__bf16)xv;
        asm volatile("s_wait_asynccnt 0x0" ::: "memory");  // y[t-1] left LDS
        __syncthreads();   // single barrier: publish x_{t+1} and h_{t+1}
    }

    // Epilogue: y[T-1] tile is in hbuf[Tlen&1]; s_endpgm's implicit wait-idle
    // guarantees the async store completes before the kernel retires.
    {
        const __bf16* hb_last = hbuf + (Tlen & 1) * (BT * HSTR);
        uint32_t ldsa = (uint32_t)(uintptr_t)(hb_last + sr * HSTR + sc);
        const __bf16* gdst = yout + ((size_t)(Tlen - 1) * Bsz + b0 + sr) * Hdim + sc;
        asm volatile("global_store_async_from_lds_b128 %0, %1, off"
                     :: "v"(gdst), "v"(ldsa) : "memory");
    }
}

// ------------------------------ layer 1 ------------------------------------
// y: [T, B, H] bf16 (from layer 0).  feats out: [B, H] f32 (= h_T).
__global__ void __launch_bounds__(256, 1)
lstm_layer1(const __bf16* __restrict__ y,
            const float* __restrict__ Wih, const float* __restrict__ Whh,
            const float* __restrict__ bih, const float* __restrict__ bhh,
            float* __restrict__ feats) {
    extern __shared__ __align__(32) char smem[];
    __bf16* wih_f = (__bf16*)(smem);                          // 131072 B
    __bf16* whh_f = (__bf16*)(smem + 131072);                 // 131072 B
    float*  bias  = (float*)(smem + 262144);                  // 2048 B
    __bf16* hbuf  = (__bf16*)(smem + 262144 + 2048);          // 2x 4352 B
    __bf16* ybuf  = (__bf16*)(smem + 262144 + 2048 + 8704);   // 2x 4352 B

    const int tid  = threadIdx.x;
    const int lane = tid & 31;
    const int w    = tid >> 5;
    const int nn   = lane & 15;
    const int hi   = lane >> 4;
    const int col  = (w << 4) + nn;
    const int b0   = blockIdx.x * BT;

    pack_weight_128(Wih, wih_f, tid);
    pack_weight_128(Whh, whh_f, tid);
    for (int i = tid; i < G4H; i += 256)           bias[i] = bih[i] + bhh[i];
    for (int i = tid; i < 2 * BT * HSTR; i += 256) hbuf[i] = (__bf16)0.0f;
    __syncthreads();

    // Hoist recurrent-path B-fragments into VGPRs (16 x v16bf = 128 VGPRs);
    // W_ih1 fragments stay in LDS (loop-invariant reads, no ordering hazard).
    v16bf bHr[4][4];
#pragma unroll
    for (int g = 0; g < 4; ++g)
#pragma unroll
        for (int kt = 0; kt < 4; ++kt)
            bHr[g][kt] = load_b_frag(whh_f, ((g * 8 + w) << 2) | kt, lane);

    const float bi_i = bias[0 * Hdim + col];
    const float bi_f = bias[1 * Hdim + col];
    const float bi_g = bias[2 * Hdim + col];
    const float bi_o = bias[3 * Hdim + col];

    // per-step y_t staging assignment: thread -> (row, 8-half group), 16B each
    const int sr = tid >> 4;
    const int sc = (tid & 15) << 3;

    // Prologue: async-stage y_0 tile into buffer 0 through the async-LDS path.
    {
        const __bf16* src = y + ((size_t)(b0 + sr)) * Hdim + sc;  // t = 0
        uint32_t ldsa = (uint32_t)(uintptr_t)(ybuf + sr * HSTR + sc);
        asm volatile("global_load_async_to_lds_b128 %0, %1, off"
                     :: "v"(ldsa), "v"(src) : "memory");
        asm volatile("s_wait_asynccnt 0x0" ::: "memory");
    }
    __syncthreads();

    v8f cst = {0.f, 0.f, 0.f, 0.f, 0.f, 0.f, 0.f, 0.f};

    for (int t = 0; t < Tlen; ++t) {
        const int cur = t & 1, nxt = cur ^ 1;
        const __bf16* yb_c = ybuf + cur * (BT * HSTR);
        __bf16*       yb_n = ybuf + nxt * (BT * HSTR);
        const __bf16* hb_c = hbuf + cur * (BT * HSTR);
        __bf16*       hb_n = hbuf + nxt * (BT * HSTR);

        // Kick off async copy of next step's activation tile (overlaps compute).
        if (t + 1 < Tlen) {
            const __bf16* src = y + ((size_t)(t + 1) * Bsz + b0 + sr) * Hdim + sc;
            uint32_t ldsa = (uint32_t)(uintptr_t)(yb_n + sr * HSTR + sc);
            asm volatile("global_load_async_to_lds_b128 %0, %1, off"
                         :: "v"(ldsa), "v"(src) : "memory");
        }

        v8f z = {0.f, 0.f, 0.f, 0.f, 0.f, 0.f, 0.f, 0.f};
        v8f acc0 = z, acc1 = z, acc2 = z, acc3 = z;

#pragma unroll
        for (int kt = 0; kt < 4; ++kt) {
            v16bf ay = load_a_frag(yb_c, HSTR, lane, kt << 5);   // input projection
            acc0 = wmma_bf16   (ay, load_b_frag(wih_f, ((0 * 8 + w) << 2) | kt, lane), acc0);
            acc1 = wmma_bf16_ra(ay, load_b_frag(wih_f, ((1 * 8 + w) << 2) | kt, lane), acc1);
            acc2 = wmma_bf16_ra(ay, load_b_frag(wih_f, ((2 * 8 + w) << 2) | kt, lane), acc2);
            acc3 = wmma_bf16_ra(ay, load_b_frag(wih_f, ((3 * 8 + w) << 2) | kt, lane), acc3);
            v16bf ah = load_a_frag(hb_c, HSTR, lane, kt << 5);   // recurrent (reg B)
            acc0 = wmma_bf16   (ah, bHr[0][kt], acc0);
            acc1 = wmma_bf16_ra(ah, bHr[1][kt], acc1);
            acc2 = wmma_bf16_ra(ah, bHr[2][kt], acc2);
            acc3 = wmma_bf16_ra(ah, bHr[3][kt], acc3);
        }

#pragma unroll
        for (int r = 0; r < 8; ++r) {
            float gi = fast_sigmoid(acc0[r] + bi_i);
            float gf = fast_sigmoid(acc1[r] + bi_f);
            float gg = fast_tanh(acc2[r] + bi_g);
            float go = fast_sigmoid(acc3[r] + bi_o);
            float cn = gf * cst[r] + gi * gg;
            cst[r] = cn;
            float h = go * fast_tanh(cn);
            int M = r + (hi << 3);
            hb_n[M * HSTR + col] = (__bf16)h;
            if (t == Tlen - 1)
                feats[(size_t)(b0 + M) * Hdim + col] = h;   // h_T = feats (f32)
        }
        asm volatile("s_wait_asynccnt 0x0" ::: "memory");  // next tile landed in LDS
        __syncthreads();                                    // publish h_{t+1}, y_{t+1}
    }
}

// ------------------------------ dense head ---------------------------------
__global__ void __launch_bounds__(256)
dense_head(const float* __restrict__ feats, const float* __restrict__ Wd,
           const float* __restrict__ bd, float* __restrict__ scores) {
    int b = threadIdx.x;
    float s = 0.f;
#pragma unroll 8
    for (int j = 0; j < Hdim; ++j) s += feats[b * Hdim + j] * Wd[j];
    scores[b] = fast_sigmoid(s + bd[0]);
}

// ---------------------------------------------------------------------------
extern "C" void kernel_launch(void* const* d_in, const int* in_sizes, int n_in,
                              void* d_out, int out_size, void* d_ws, size_t ws_size,
                              hipStream_t stream) {
    const float* x    = (const float*)d_in[0];
    // d_in[1] = seq_lengths (unused by the reference computation)
    const float* Wih0 = (const float*)d_in[2];
    const float* Whh0 = (const float*)d_in[3];
    const float* bih0 = (const float*)d_in[4];
    const float* bhh0 = (const float*)d_in[5];
    const float* Wih1 = (const float*)d_in[6];
    const float* Whh1 = (const float*)d_in[7];
    const float* bih1 = (const float*)d_in[8];
    const float* bhh1 = (const float*)d_in[9];
    const float* Wd   = (const float*)d_in[10];
    const float* bd   = (const float*)d_in[11];

    float* scores = (float*)d_out;            // [256]
    float* feats  = (float*)d_out + Bsz;      // [256,128]
    __bf16* ybuf  = (__bf16*)d_ws;            // [T,B,H] bf16 = 128 MB

    const size_t smemA = 32768 + 131072 + 2048 + 2 * BT * HSTR * 2 + 2 * BT * XSTR * 2;
    const size_t smemB = 131072 + 131072 + 2048 + 2 * BT * HSTR * 2 + 2 * BT * HSTR * 2;

    (void)hipFuncSetAttribute(reinterpret_cast<const void*>(lstm_layer0),
                              hipFuncAttributeMaxDynamicSharedMemorySize, (int)smemA);
    (void)hipFuncSetAttribute(reinterpret_cast<const void*>(lstm_layer1),
                              hipFuncAttributeMaxDynamicSharedMemorySize, (int)smemB);

    lstm_layer0<<<Bsz / BT, 256, smemA, stream>>>(x, Wih0, Whh0, bih0, bhh0, ybuf);
    lstm_layer1<<<Bsz / BT, 256, smemB, stream>>>(ybuf, Wih1, Whh1, bih1, bhh1, feats);
    dense_head<<<1, 256, 0, stream>>>(feats, Wd, bd, scores);
}
